// MultiHeadAttention_87205015978135
// MI455X (gfx1250) — compile-verified, matrix operand order checked by reading, not run on hardware
//
#include <hip/hip_runtime.h>

// ---------------------------------------------------------------------------
// MultiHeadAttention forward for MI455X (gfx1250, wave32, WMMA bf16)
//   B=2, T=2048, C=1024, H=16, D=64
//   - bf16 WMMA for all four GEMM stages + streaming-softmax attention
//   - GEMM: 4-wave workgroups, W-tile double-buffered in LDS via
//     global_load_async_to_lds_b128 (ASYNCcnt) -- CDNA5 async path
// ---------------------------------------------------------------------------

typedef __bf16 bf16_t;
typedef __attribute__((ext_vector_type(16))) __bf16 v16bf;
typedef __attribute__((ext_vector_type(8)))  __bf16 v8bf;
typedef __attribute__((ext_vector_type(8)))  float  v8f;

#define T_SEQ   2048
#define C_DIM   1024
#define NHEAD   16
#define HDIM    64
#define MTOT    4096          // B * T
#define ATT_SCALE 0.125f      // 1/sqrt(64)

// ----- helpers --------------------------------------------------------------

__device__ __forceinline__ bf16_t f2bf(float f) {
  unsigned u = __builtin_bit_cast(unsigned, f);
  unsigned r = u + 0x7FFFu + ((u >> 16) & 1u);   // round-to-nearest-even
  unsigned short h = (unsigned short)(r >> 16);
  return __builtin_bit_cast(bf16_t, h);
}

// WMMA D = A(16x32 bf16) * B(32x16 bf16) + C(16x16 f32)
__device__ __forceinline__ v8f wmma_bf16(v16bf a, v16bf b, v8f c) {
  return __builtin_amdgcn_wmma_f32_16x16x32_bf16(
      /*neg_a=*/false, a, /*neg_b=*/false, b,
      /*c_mod=*/(short)0, c, /*reuse_a=*/false, /*reuse_b=*/false);
}

// CDNA5 async copy: 16 bytes global -> LDS, tracked by ASYNCcnt.
// lds_off = LDS byte address (low 32 bits of the generic pointer).
__device__ __forceinline__ void async_copy_b128(unsigned lds_off,
                                                const bf16_t* gsrc) {
  unsigned long long ga = (unsigned long long)(uintptr_t)gsrc;
  asm volatile("global_load_async_to_lds_b128 %0, %1, off"
               :: "v"(lds_off), "v"(ga) : "memory");
}

__device__ __forceinline__ void wait_asynccnt0() {
  asm volatile("s_wait_asynccnt 0x0" ::: "memory");
}

// Load a 16x32 bf16 fragment (ISA 7.12.2 16-bit A layout) from a row-major
// matrix: rows r0..r0+15, cols c0..c0+31, leading dim ld (elements).
// lanes 0-15 hold (row = r0+lane) K = 0-7 and 16-23; lanes 16-31 hold K = 8-15
// and 24-31.  The SAME pattern serves as the B operand when the B matrix is
// stored column-contiguous (i.e. memory is [N][K] row-major).
__device__ __forceinline__ v16bf load_frag(const bf16_t* base, int ld,
                                           int r0, int c0) {
  int lane = threadIdx.x & 31;
  int nn   = lane & 15;
  int half = lane >> 4;
  const bf16_t* p = base + (size_t)(r0 + nn) * ld + c0 + (half << 3);
  v8bf lo = *reinterpret_cast<const v8bf*>(p);        // K = half*8 .. +8
  v8bf hi = *reinterpret_cast<const v8bf*>(p + 16);   // K = 16+half*8 .. +8
  return __builtin_shufflevector(lo, hi, 0, 1, 2, 3, 4, 5, 6, 7,
                                 8, 9, 10, 11, 12, 13, 14, 15);
}

__device__ __forceinline__ float rmax16(float v) {
  #pragma unroll
  for (int off = 1; off < 16; off <<= 1)
    v = fmaxf(v, __shfl_xor(v, off, 16));
  return v;
}

__device__ __forceinline__ float rsum16(float v) {
  #pragma unroll
  for (int off = 1; off < 16; off <<= 1)
    v += __shfl_xor(v, off, 16);
  return v;
}

// ----- kernel 1: f32 -> bf16 conversion ------------------------------------

__global__ void cvt_f32_bf16(const float* __restrict__ in,
                             bf16_t* __restrict__ out, int n) {
  int i = blockIdx.x * blockDim.x + threadIdx.x;
  if (i < n) out[i] = f2bf(in[i]);
}

// ----- kernel 2: GEMM  Y = A(4096x1024) @ W^T(1024x1024) + bias ------------
// 4 waves / workgroup; each wave owns a 32(M) x 64(N) tile, all waves share
// the same N block.  Per K-step (32): the 64x32 W tile (4 KB) is staged into
// double-buffered LDS with global_load_async_to_lds_b128; A fragments load
// directly from global (b128) so VMEM overlaps the DS+WMMA stream.
// mode 0: store bf16 row-major [M][N], value scaled by `scale`
// mode 1: store bf16 transposed for V: out[(b*1024+n)*2048 + t]
// mode 2: store f32 row-major [M][N]

__global__ void gemm_proj(const bf16_t* __restrict__ A,
                          const bf16_t* __restrict__ W,
                          const float* __restrict__ bias,
                          void* __restrict__ out, int mode, float scale) {
  __shared__ alignas(16) bf16_t ldsB[2][64 * 32];

  int tid  = threadIdx.x;            // 0..127
  int wave = tid >> 5;
  int lane = tid & 31;
  int nn = lane & 15, half = lane >> 4;
  int m0 = blockIdx.x * 128 + wave * 32;
  int n0 = blockIdx.y * 64;

  // stage 64 rows x 32 cols of W (rows n0.., cols k..) into ldsB[buf]
  auto stage = [&](int buf, int k) {
    #pragma unroll
    for (int c = 0; c < 2; ++c) {
      int chunk = tid * 2 + c;        // 0..255, 16B each
      int row = chunk >> 2;           // 0..63
      int col = (chunk & 3) * 8;      // 0,8,16,24 (elements)
      const bf16_t* src = W + (size_t)(n0 + row) * C_DIM + k + col;
      unsigned dst = (unsigned)(uintptr_t)&ldsB[buf][row * 32 + col];
      async_copy_b128(dst, src);
    }
  };

  v8f c[2][4];
  #pragma unroll
  for (int i = 0; i < 2; ++i)
    #pragma unroll
    for (int j = 0; j < 4; ++j)
      c[i][j] = v8f{0.f, 0.f, 0.f, 0.f, 0.f, 0.f, 0.f, 0.f};

  stage(0, 0);
  int buf = 0;
  for (int k = 0; k < C_DIM; k += 32) {
    v16bf a0 = load_frag(A, C_DIM, m0,      k);
    v16bf a1 = load_frag(A, C_DIM, m0 + 16, k);
    wait_asynccnt0();          // our slice of ldsB[buf] landed
    __syncthreads();           // everyone's slice landed
    if (k + 32 < C_DIM) stage(buf ^ 1, k + 32);   // prefetch next tile
    #pragma unroll
    for (int j = 0; j < 4; ++j) {
      v16bf b = load_frag(&ldsB[buf][0], 32, j * 16, 0);
      c[0][j] = wmma_bf16(a0, b, c[0][j]);
      c[1][j] = wmma_bf16(a1, b, c[1][j]);
    }
    buf ^= 1;
  }

  #pragma unroll
  for (int j = 0; j < 4; ++j) {
    int n = n0 + j * 16 + nn;
    float bv = bias[n];
    #pragma unroll
    for (int i = 0; i < 2; ++i) {
      #pragma unroll
      for (int r = 0; r < 8; ++r) {
        int m = m0 + i * 16 + r + 8 * half;
        float val = (c[i][j][r] + bv) * scale;
        if (mode == 0) {
          ((bf16_t*)out)[(size_t)m * C_DIM + n] = f2bf(val);
        } else if (mode == 1) {
          int b = m >> 11, t = m & 2047;                 // m = b*2048 + t
          ((bf16_t*)out)[((size_t)(b * 1024 + n)) * T_SEQ + t] = f2bf(val);
        } else {
          ((float*)out)[(size_t)m * C_DIM + n] = val;
        }
      }
    }
  }
}

// ----- kernel 3: streaming-softmax attention --------------------------------
// one wave per block: 16 query rows of one (b, h).
// Q pre-scaled by 1/sqrt(d); loop over key tiles of 32, online softmax,
// P round-trips through LDS (C-layout -> A-fragment layout, bf16),
// out accumulated as 16x64 f32 via 4 WMMAs per key tile.

__global__ void attn_flash(const bf16_t* __restrict__ Qb,
                           const bf16_t* __restrict__ Kb,
                           const bf16_t* __restrict__ Vt,
                           const float* __restrict__ mask,
                           bf16_t* __restrict__ AO) {
  __shared__ alignas(16) bf16_t ldsP[16 * 32];

  int qt = blockIdx.x;                 // query tile (0..127)
  int bh = blockIdx.y;                 // b*16 + h
  int b = bh >> 4, h = bh & 15;
  int lane = threadIdx.x & 31;
  int nn = lane & 15, half = lane >> 4;

  int qrow0 = b * T_SEQ + qt * 16;
  v16bf aq0 = load_frag(Qb, C_DIM, qrow0, h * HDIM);
  v16bf aq1 = load_frag(Qb, C_DIM, qrow0, h * HDIM + 32);

  v8f c[4];
  #pragma unroll
  for (int j = 0; j < 4; ++j)
    c[j] = v8f{0.f, 0.f, 0.f, 0.f, 0.f, 0.f, 0.f, 0.f};

  float mrun[8], lrun[8];
  #pragma unroll
  for (int r = 0; r < 8; ++r) { mrun[r] = -3.0e38f; lrun[r] = 0.f; }

  const bf16_t* VtB = Vt + (size_t)b * 1024 * T_SEQ;  // [n=h*64+d][t]
  const float* mbase = mask + (size_t)(qt * 16) * T_SEQ;

  for (int kbase = 0; kbase < T_SEQ; kbase += 32) {
    // ---- scores: S(16x32) = Q(16x64) @ K^T, two 16x16 sub-tiles ----
    v8f s0 = v8f{0.f, 0.f, 0.f, 0.f, 0.f, 0.f, 0.f, 0.f};
    v8f s1 = v8f{0.f, 0.f, 0.f, 0.f, 0.f, 0.f, 0.f, 0.f};
    {
      v16bf k00 = load_frag(Kb, C_DIM, b * T_SEQ + kbase,      h * HDIM);
      v16bf k01 = load_frag(Kb, C_DIM, b * T_SEQ + kbase,      h * HDIM + 32);
      s0 = wmma_bf16(aq0, k00, s0);
      s0 = wmma_bf16(aq1, k01, s0);
      v16bf k10 = load_frag(Kb, C_DIM, b * T_SEQ + kbase + 16, h * HDIM);
      v16bf k11 = load_frag(Kb, C_DIM, b * T_SEQ + kbase + 16, h * HDIM + 32);
      s1 = wmma_bf16(aq0, k10, s1);
      s1 = wmma_bf16(aq1, k11, s1);
    }

    // ---- additive mask (2D [Tq][Tk]) ----
    #pragma unroll
    for (int r = 0; r < 8; ++r) {
      const float* mp = mbase + (size_t)(r + 8 * half) * T_SEQ + kbase;
      s0[r] += mp[nn];
      s1[r] += mp[16 + nn];
    }

    // ---- online softmax update + write P tile (bf16) to LDS ----
    #pragma unroll
    for (int r = 0; r < 8; ++r) {
      float tm = rmax16(fmaxf(s0[r], s1[r]));
      float nm = fmaxf(mrun[r], tm);
      float al = __expf(mrun[r] - nm);
      float p0 = __expf(s0[r] - nm);
      float p1 = __expf(s1[r] - nm);
      float rs = rsum16(p0 + p1);
      lrun[r] = lrun[r] * al + rs;
      mrun[r] = nm;
      #pragma unroll
      for (int j = 0; j < 4; ++j) c[j][r] *= al;
      int row = r + 8 * half;
      ldsP[row * 32 + nn]      = f2bf(p0);
      ldsP[row * 32 + 16 + nn] = f2bf(p1);
    }
    // single wave: DS ops are in-order; compiler inserts s_wait_dscnt.

    // ---- out(16x64) += P(16x32) @ V(32x64)  (V stored transposed) ----
    v16bf pa = load_frag(ldsP, 32, 0, 0);
    #pragma unroll
    for (int j = 0; j < 4; ++j) {
      v16bf vb = load_frag(VtB, T_SEQ, h * HDIM + j * 16, kbase);
      c[j] = wmma_bf16(pa, vb, c[j]);
    }
  }

  // ---- normalize and store to AO (bf16 row-major [4096][1024]) ----
  #pragma unroll
  for (int r = 0; r < 8; ++r) {
    float inv = 1.0f / lrun[r];
    #pragma unroll
    for (int j = 0; j < 4; ++j) c[j][r] *= inv;
  }
  #pragma unroll
  for (int j = 0; j < 4; ++j) {
    #pragma unroll
    for (int r = 0; r < 8; ++r) {
      int m = qrow0 + r + 8 * half;
      int n = h * HDIM + j * 16 + nn;
      AO[(size_t)m * C_DIM + n] = f2bf(c[j][r]);
    }
  }
}

// ---------------------------------------------------------------------------

extern "C" void kernel_launch(void* const* d_in, const int* in_sizes, int n_in,
                              void* d_out, int out_size, void* d_ws,
                              size_t ws_size, hipStream_t stream) {
  const float* x    = (const float*)d_in[0];
  const float* mask = (const float*)d_in[1];
  const float* Wq = (const float*)d_in[2];  const float* bq = (const float*)d_in[3];
  const float* Wk = (const float*)d_in[4];  const float* bk = (const float*)d_in[5];
  const float* Wv = (const float*)d_in[6];  const float* bv = (const float*)d_in[7];
  const float* Wo = (const float*)d_in[8];  const float* bo = (const float*)d_in[9];

  char* ws = (char*)d_ws;
  const size_t MiB = 1024 * 1024;
  bf16_t* xb  = (bf16_t*)(ws + 0);          //  8 MiB  x  bf16 [4096][1024]
  bf16_t* wqb = (bf16_t*)(ws + 8  * MiB);   //  2 MiB
  bf16_t* wkb = (bf16_t*)(ws + 10 * MiB);   //  2 MiB
  bf16_t* wvb = (bf16_t*)(ws + 12 * MiB);   //  2 MiB
  bf16_t* wob = (bf16_t*)(ws + 14 * MiB);   //  2 MiB
  bf16_t* Qb  = (bf16_t*)(ws + 16 * MiB);   //  8 MiB  (pre-scaled by 1/sqrt d)
  bf16_t* Kb  = (bf16_t*)(ws + 24 * MiB);   //  8 MiB
  bf16_t* Vt  = (bf16_t*)(ws + 32 * MiB);   //  8 MiB  V transposed [b][n][t]
  bf16_t* AO  = (bf16_t*)(ws + 40 * MiB);   //  8 MiB  attention out bf16

  // 1) fp32 -> bf16 staging
  const int NX = MTOT * C_DIM;      // 4194304
  const int NW = C_DIM * C_DIM;     // 1048576
  cvt_f32_bf16<<<(NX + 255) / 256, 256, 0, stream>>>(x,  xb,  NX);
  cvt_f32_bf16<<<(NW + 255) / 256, 256, 0, stream>>>(Wq, wqb, NW);
  cvt_f32_bf16<<<(NW + 255) / 256, 256, 0, stream>>>(Wk, wkb, NW);
  cvt_f32_bf16<<<(NW + 255) / 256, 256, 0, stream>>>(Wv, wvb, NW);
  cvt_f32_bf16<<<(NW + 255) / 256, 256, 0, stream>>>(Wo, wob, NW);

  // 2) Q/K/V projections (WMMA bf16, async-LDS W staging)
  dim3 gg(MTOT / 128, C_DIM / 64);  // 32 x 16
  dim3 bw(128);                     // 4 waves
  gemm_proj<<<gg, bw, 0, stream>>>(xb, wqb, bq, (void*)Qb, 0, ATT_SCALE);
  gemm_proj<<<gg, bw, 0, stream>>>(xb, wkb, bk, (void*)Kb, 0, 1.0f);
  gemm_proj<<<gg, bw, 0, stream>>>(xb, wvb, bv, (void*)Vt, 1, 1.0f);

  // 3) streaming-softmax attention (WMMA bf16)
  attn_flash<<<dim3(T_SEQ / 16, 2 * NHEAD), dim3(32), 0, stream>>>(
      Qb, Kb, Vt, mask, AO);

  // 4) output projection, f32 result
  gemm_proj<<<gg, bw, 0, stream>>>(AO, wob, bo, d_out, 2, 1.0f);
}
